// MBConv_82308753260765
// MI455X (gfx1250) — matrix operand order repaired and checked by reading, not compiled, not run on hardware
//
#include <hip/hip_runtime.h>
#include <hip/hip_bf16.h>
#include <math.h>

typedef __attribute__((ext_vector_type(16))) __bf16 v16bf;
typedef __attribute__((ext_vector_type(8)))  float  v8f;

#define EPSV 1e-5f
#define NSPAT 65536          // 16^4 spatial points
#define HID   128
#define CIN   32

__device__ inline float gelu_f(float v) {
    return 0.5f * v * (1.0f + erff(v * 0.70710678118654752f));
}

__device__ inline unsigned short f2bf(float f) {
    unsigned u = __float_as_uint(f);
    u += 0x7FFFu + ((u >> 16) & 1u);     // round-to-nearest-even
    return (unsigned short)(u >> 16);
}

template <int BS>
__device__ inline float block_sum(float v, float* red) {
    int t = threadIdx.x;
    red[t] = v;
    __syncthreads();
    for (int s = BS / 2; s > 0; s >>= 1) {
        if (t < s) red[t] += red[t + s];
        __syncthreads();
    }
    float r = red[0];
    __syncthreads();
    return r;
}

// ---------------------------------------------------------------- init stats
__global__ void k_init(float* stats) {
    int i = threadIdx.x;
    if (i < 264) stats[i] = 0.0f;      // 8 stats + 128 chsum + 128 y
}

// ------------------------------------------------------- global sum / sumsq
__global__ __launch_bounds__(256) void k_stats(const float* __restrict__ x,
                                               int n, float* __restrict__ out2) {
    __shared__ float red[256];
    float s = 0.f, q = 0.f;
    for (int i = blockIdx.x * 256 + threadIdx.x; i < n; i += gridDim.x * 256) {
        float v = x[i];
        s += v; q += v * v;
    }
    float bs = block_sum<256>(s, red);
    float bq = block_sum<256>(q, red);
    if (threadIdx.x == 0) {
        atomicAdd(out2 + 0, bs);
        atomicAdd(out2 + 1, bq);
    }
}

// ------------------------------------------- GN0 folded into conv1 (32->128)
__global__ __launch_bounds__(256) void k_conv1(const float* __restrict__ x,
                                               const float* __restrict__ g0w,
                                               const float* __restrict__ g0b,
                                               const float* __restrict__ w1,
                                               const float* __restrict__ stats,
                                               float* __restrict__ h1,
                                               float* __restrict__ s1out) {
    __shared__ float W[HID * CIN];
    __shared__ float Ac[CIN], Bc[CIN];
    __shared__ float red[256];
    const int tid = threadIdx.x;
    const float ninv = 1.0f / 2097152.0f;
    float mu  = stats[0] * ninv;
    float var = stats[1] * ninv - mu * mu;
    float inv = rsqrtf(var + EPSV);
    for (int i = tid; i < HID * CIN; i += 256) W[i] = w1[i];
    if (tid < CIN) {
        Ac[tid] = inv * g0w[tid];
        Bc[tid] = g0b[tid] - mu * inv * g0w[tid];
    }
    __syncthreads();

    const int p = blockIdx.x * 256 + tid;   // 256 blocks -> 65536 points
    float xn[CIN];
#pragma unroll
    for (int c = 0; c < CIN; ++c)
        xn[c] = x[c * NSPAT + p] * Ac[c] + Bc[c];

    float ssum = 0.f, ssq = 0.f;
    for (int h = 0; h < HID; ++h) {
        float acc = 0.f;
#pragma unroll
        for (int c = 0; c < CIN; ++c) acc += W[h * CIN + c] * xn[c];
        h1[(size_t)h * NSPAT + p] = acc;
        ssum += acc; ssq += acc * acc;
    }
    float bs = block_sum<256>(ssum, red);
    float bq = block_sum<256>(ssq, red);
    if (tid == 0) { atomicAdd(s1out + 0, bs); atomicAdd(s1out + 1, bq); }
}

// ------------------------------------- GN1 + GELU + convert to bf16 activ.
__global__ __launch_bounds__(256) void k_gn1_gelu_bf16(const float* __restrict__ h1,
                                                       const float* __restrict__ stats,
                                                       const float* __restrict__ gw,
                                                       const float* __restrict__ gb,
                                                       unsigned short* __restrict__ actbf) {
    const float ninv = 1.0f / 8388608.0f;
    float mu  = stats[2] * ninv;
    float var = stats[3] * ninv - mu * mu;
    float inv = rsqrtf(var + EPSV);
    int i = blockIdx.x * 256 + threadIdx.x;     // 32768 blocks
    int c = i >> 16;
    float v = (h1[i] - mu) * inv * gw[c] + gb[c];
    actbf[i] = f2bf(gelu_f(v));
}

// --------------------------- pack w2 fp32[o][i][81] -> bf16 [t][ib][o][c32]
__global__ __launch_bounds__(256) void k_pack_w2(const float* __restrict__ w2,
                                                 unsigned short* __restrict__ w2bf) {
    int d = blockIdx.x * 256 + threadIdx.x;     // 5184 blocks = 1327104
    if (d >= 81 * 4 * 128 * 32) return;
    int c  = d & 31;
    int o  = (d >> 5) & 127;
    int ib = (d >> 12) & 3;
    int t  = d >> 14;
    int i  = ib * 32 + c;
    w2bf[d] = f2bf(w2[((size_t)o * 128 + i) * 81 + t]);
}

// ------------------------------------------------ conv2: 3^4 implicit GEMM
// grid 256 (one per (x,y)), block 256 = 8 waves; wave w -> 16 out-channels.
// 8 z-planes per pass: one A (weight) fragment feeds 8 WMMAs.
// W axis staged with 18 padded slots (slot = w+1, slots 0/17 zero) so every
// B fragment is an unconditional pair of aligned ds_load_b128.
union FragU { uint4 u[2]; v16bf v; };

#define ZG 8                              // z-planes per accumulation group
#define PZ (ZG + 2)                       // staged z-planes (halo)
#define WS 18                             // padded W slots

__global__ __launch_bounds__(256) void k_conv2_wmma(const unsigned short* __restrict__ actbf,
                                                    const unsigned short* __restrict__ w2bf,
                                                    float* __restrict__ h2,
                                                    float* __restrict__ stats) {
    __shared__ unsigned short sAct[9 * PZ * WS * 32];   // [dxy][pz][slot][c] ~101 KB
    __shared__ float red[256];

    const int tid   = threadIdx.x;
    const int lane  = tid & 31;
    const int wavi  = tid >> 5;          // 0..7 : o-tile
    const int X     = blockIdx.x >> 4;
    const int Y     = blockIdx.x & 15;
    const int n     = lane & 15;         // output w within tile (= N column)
    const int khalf = lane >> 4;         // 0/1 -> K-half of fragments

    float ssum = 0.f, ssq = 0.f;

    for (int zg = 0; zg < 16 / ZG; ++zg) {
        const int Zb = zg * ZG;
        v8f acc[ZG];
#pragma unroll
        for (int zi = 0; zi < ZG; ++zi) acc[zi] = v8f{};

        for (int ib = 0; ib < 4; ++ib) {
            __syncthreads();
            // ---- stage 32 ch x 9 (dx,dy) x PZ z-planes x 16 W (+2 pad)
            for (int r = tid; r < 32 * 9 * PZ; r += 256) {
                int ic   = r & 31;
                int rest = r >> 5;               // 0..(9*PZ-1)
                int dxy  = rest % 9;
                int pz   = rest / 9;             // 0..PZ-1
                int xx = X + dxy / 3 - 1;
                int yy = Y + dxy % 3 - 1;
                int zz = Zb + pz - 1;
                union { uint4 q[2]; unsigned short s[16]; } row;
                row.q[0] = make_uint4(0, 0, 0, 0);
                row.q[1] = make_uint4(0, 0, 0, 0);
                if ((unsigned)xx < 16u && (unsigned)yy < 16u && (unsigned)zz < 16u) {
                    const uint4* g = (const uint4*)(actbf +
                        (size_t)(ib * 32 + ic) * NSPAT + xx * 4096 + yy * 256 + zz * 16);
                    row.q[0] = g[0];
                    row.q[1] = g[1];
                }
                unsigned short* dst = &sAct[(dxy * PZ + pz) * (WS * 32) + ic];
                dst[0]       = 0;                // slot 0   (wsrc = -1)
                dst[17 * 32] = 0;                // slot 17  (wsrc = 16)
#pragma unroll
                for (int w = 0; w < 16; ++w) dst[(w + 1) * 32] = row.s[w];
            }
            __syncthreads();

            // ---- 27 (dx,dy,dz) x 3 kw taps; K-chunk = 32 input channels
            for (int t27 = 0; t27 < 27; ++t27) {
                const int dxy = t27 / 3;         // kx*3+ky
                const int kz  = t27 % 3;
                // base LDS address: slot index n, zi/kw add constant offsets
                const unsigned short* bbase =
                    &sAct[((dxy * PZ + kz) * WS + n) * 32 + khalf * 16];
#pragma unroll
                for (int kw = 0; kw < 3; ++kw) {
                    const int t = t27 * 3 + kw;
                    // A fragment: [t][ib][o][c32]; 16-bit 16x32 A layout
                    FragU a;
                    const uint4* ap = (const uint4*)(w2bf +
                        ((size_t)(t * 4 + ib) * 128 + wavi * 16 + (lane & 15)) * 32 +
                        khalf * 16);
                    a.u[0] = ap[0];
                    a.u[1] = ap[1];
#pragma unroll
                    for (int zi = 0; zi < ZG; ++zi) {
                        FragU b;
                        const uint4* bp = (const uint4*)
                            (bbase + zi * (WS * 32) + kw * 32);
                        b.u[0] = bp[0];
                        b.u[1] = bp[1];
                        acc[zi] = __builtin_amdgcn_wmma_f32_16x16x32_bf16(
                            false, a.v, false, b.v, (short)0, acc[zi], false, false);
                    }
                }
            }
        }
        // ---- write ZG 16x16 tiles: VGPR r -> M = r + 8*khalf, N = lane&15
        const int obase = wavi * 16 + khalf * 8;
#pragma unroll
        for (int zi = 0; zi < ZG; ++zi) {
            const int sp = X * 256 + Y * 16 + (Zb + zi);
#pragma unroll
            for (int r = 0; r < 8; ++r) {
                float v = acc[zi][r];
                h2[(size_t)(obase + r) * NSPAT + sp * 16 + n] = v;
                ssum += v; ssq += v * v;
            }
        }
    }
    float bs = block_sum<256>(ssum, red);
    float bq = block_sum<256>(ssq, red);
    if (tid == 0) { atomicAdd(stats + 4, bs); atomicAdd(stats + 5, bq); }
}

// --------------------------- GN2 + GELU in place + per-channel sums for SE
__global__ __launch_bounds__(256) void k_gn2_gelu(float* __restrict__ h2,
                                                  const float* __restrict__ stats,
                                                  const float* __restrict__ gw,
                                                  const float* __restrict__ gb,
                                                  float* __restrict__ chsum) {
    __shared__ float red[256];
    const float ninv = 1.0f / 8388608.0f;
    float mu  = stats[4] * ninv;
    float var = stats[5] * ninv - mu * mu;
    float inv = rsqrtf(var + EPSV);
    int c = blockIdx.y;
    size_t i = (size_t)c * NSPAT + blockIdx.x * 256 + threadIdx.x;
    float v = (h2[i] - mu) * inv * gw[c] + gb[c];
    float g = gelu_f(v);
    h2[i] = g;
    float bs = block_sum<256>(g, red);
    if (threadIdx.x == 0) atomicAdd(chsum + c, bs);
}

// ----------------------------------------------------------------- SE gate
__global__ void k_se(const float* __restrict__ chsum,
                     const float* __restrict__ w1se,   // [8][128]
                     const float* __restrict__ w2se,   // [128][8]
                     float* __restrict__ y) {
    __shared__ float m[128];
    __shared__ float t1[8];
    int tid = threadIdx.x;                 // 128 threads
    m[tid] = chsum[tid] * (1.0f / 65536.0f);
    __syncthreads();
    if (tid < 8) {
        float a = 0.f;
        for (int c = 0; c < 128; ++c) a += w1se[tid * 128 + c] * m[c];
        t1[tid] = gelu_f(a);
    }
    __syncthreads();
    float a = 0.f;
#pragma unroll
    for (int j = 0; j < 8; ++j) a += w2se[tid * 8 + j] * t1[j];
    y[tid] = 1.0f / (1.0f + expf(-a));
}

// -------------------------------------- conv3 (128->32) with SE gate folded
__global__ __launch_bounds__(256) void k_conv3(const float* __restrict__ h2g,
                                               const float* __restrict__ w3,
                                               const float* __restrict__ y,
                                               float* __restrict__ h3,
                                               float* __restrict__ s3out) {
    __shared__ float W[32 * 128];
    __shared__ float red[256];
    const int tid = threadIdx.x;
    for (int i = tid; i < 32 * 128; i += 256) W[i] = w3[i] * y[i & 127];
    __syncthreads();

    const int p = blockIdx.x * 256 + tid;    // 256 blocks
    float acc[32];
#pragma unroll
    for (int o = 0; o < 32; ++o) acc[o] = 0.f;
    for (int c = 0; c < 128; ++c) {
        float hv = h2g[(size_t)c * NSPAT + p];
#pragma unroll
        for (int o = 0; o < 32; ++o) acc[o] += W[o * 128 + c] * hv;
    }
    float ssum = 0.f, ssq = 0.f;
#pragma unroll
    for (int o = 0; o < 32; ++o) {
        h3[(size_t)o * NSPAT + p] = acc[o];
        ssum += acc[o]; ssq += acc[o] * acc[o];
    }
    float bs = block_sum<256>(ssum, red);
    float bq = block_sum<256>(ssq, red);
    if (tid == 0) { atomicAdd(s3out + 0, bs); atomicAdd(s3out + 1, bq); }
}

// ------------------------------------------------------------- GN3 -> out
__global__ __launch_bounds__(256) void k_gn3(const float* __restrict__ h3,
                                             const float* __restrict__ stats,
                                             const float* __restrict__ gw,
                                             const float* __restrict__ gb,
                                             float* __restrict__ out) {
    const float ninv = 1.0f / 2097152.0f;
    float mu  = stats[6] * ninv;
    float var = stats[7] * ninv - mu * mu;
    float inv = rsqrtf(var + EPSV);
    int i = blockIdx.x * 256 + threadIdx.x;   // 8192 blocks
    int o = i >> 16;
    out[i] = (h3[i] - mu) * inv * gw[o] + gb[o];
}

// ---------------------------------------------------------------- launcher
extern "C" void kernel_launch(void* const* d_in, const int* in_sizes, int n_in,
                              void* d_out, int out_size, void* d_ws, size_t ws_size,
                              hipStream_t stream) {
    const float* x    = (const float*)d_in[0];
    const float* g0w  = (const float*)d_in[1];
    const float* g0b  = (const float*)d_in[2];
    const float* w1   = (const float*)d_in[3];
    const float* gn1w = (const float*)d_in[4];
    const float* gn1b = (const float*)d_in[5];
    const float* w2   = (const float*)d_in[6];
    const float* gn2w = (const float*)d_in[7];
    const float* gn2b = (const float*)d_in[8];
    const float* sew1 = (const float*)d_in[9];
    const float* sew2 = (const float*)d_in[10];
    const float* w3   = (const float*)d_in[11];
    const float* gn3w = (const float*)d_in[12];
    const float* gn3b = (const float*)d_in[13];
    float* out = (float*)d_out;

    // workspace carve-up (all 16B aligned)
    float*          stats = (float*)d_ws;                 // [0..7] GN stats
    float*          chsum = stats + 8;                    // [128]
    float*          yv    = stats + 136;                  // [128]
    float*          bufA  = stats + 256;                  // h1 then h2 (32 MB)
    unsigned short* actbf = (unsigned short*)(bufA + (size_t)HID * NSPAT);
    unsigned short* w2bf  = actbf + (size_t)HID * NSPAT;  // 1327104 bf16
    float*          h3    = (float*)(w2bf + 81 * 4 * 128 * 32);

    k_init<<<1, 256, 0, stream>>>(stats);
    k_stats<<<2048, 256, 0, stream>>>(x, 2097152, stats);
    k_conv1<<<256, 256, 0, stream>>>(x, g0w, g0b, w1, stats, bufA, stats + 2);
    k_gn1_gelu_bf16<<<32768, 256, 0, stream>>>(bufA, stats, gn1w, gn1b, actbf);
    k_pack_w2<<<5184, 256, 0, stream>>>(w2, w2bf);
    k_conv2_wmma<<<256, 256, 0, stream>>>(actbf, w2bf, bufA, stats);
    k_gn2_gelu<<<dim3(256, 128), 256, 0, stream>>>(bufA, stats, gn2w, gn2b, chsum);
    k_se<<<1, 128, 0, stream>>>(chsum, sew1, sew2, yv);
    k_conv3<<<256, 256, 0, stream>>>(bufA, w3, yv, h3, stats + 6);
    k_gn3<<<8192, 256, 0, stream>>>(h3, stats, gn3w, gn3b, out);
}
